// ToFace_Unet_Dwt_locate_38740605010228
// MI455X (gfx1250) — compile-verified, matrix operand order checked by reading, not run on hardware
//
#include <hip/hip_runtime.h>

typedef __attribute__((ext_vector_type(16))) _Float16 v16h;
typedef __attribute__((ext_vector_type(8)))  _Float16 v8h;
typedef __attribute__((ext_vector_type(8)))  float    v8f;
typedef __attribute__((ext_vector_type(4)))  float    v4f;

// db2 filters, reversed for cross-correlation (jax lax.conv = correlation)
#define H0_0  0.48296291314469025f
#define H0_1  0.836516303737469f
#define H0_2  0.22414386804185735f
#define H0_3 -0.12940952255092145f
#define H1_0  0.12940952255092145f
#define H1_1 -0.22414386804185735f
#define H1_2  0.836516303737469f
#define H1_3 -0.48296291314469025f

// ---------------------------------------------------------------------------
// Kernel 1: fused double channel-DWT. One thread per (b,h,w) pixel, streams
// 360 channels with 4-tap rolling windows: DWT1-hi -> DWT2-lo/hi.
// Reads coalesced over W; x is a pure 94MB stream -> non-temporal loads.
// cd2 written in exact [B,180,H,W] flat order (reshape semantics need it).
// ---------------------------------------------------------------------------
__global__ __launch_bounds__(256) void dwt2_kernel(const float* __restrict__ x,
                                                   float* __restrict__ cd2) {
  int p  = blockIdx.x * 256 + threadIdx.x;   // 0..65535 pixels
  int b  = p >> 12;                          // /4096
  int hw = p & 4095;
  const float* __restrict__ xin  = x   + ((size_t)b * 360) * 4096 + hw;
  float*       __restrict__ outp = cd2 + ((size_t)b * 180) * 4096 + hw;

  float xb0 = 0.f, xb1 = 0.f, xb2 = 0.f, xb3 = 0.f;  // zero-padded x window
  float db0 = 0.f, db1 = 0.f, db2 = 0.f, db3 = 0.f;  // zero-padded d1 window

  #pragma unroll 4
  for (int c = 0; c < 360; ++c) {
    float xv = __builtin_nontemporal_load(&xin[(size_t)c * 4096]);
    xb0 = xb1; xb1 = xb2; xb2 = xb3; xb3 = xv;
    if (c & 1) {                             // d1[k] ready, k = (c-1)/2
      int k = (c - 1) >> 1;
      float d1 = H1_0*xb0 + H1_1*xb1 + H1_2*xb2 + H1_3*xb3;
      db0 = db1; db1 = db2; db2 = db3; db3 = d1;
      if (k & 1) {                           // d2[m] ready, m = (k-1)/2
        int m = (k - 1) >> 1;
        float lo = H0_0*db0 + H0_1*db1 + H0_2*db2 + H0_3*db3;
        float hi = H1_0*db0 + H1_1*db1 + H1_2*db2 + H1_3*db3;
        outp[(size_t)m        * 4096] = lo;
        outp[(size_t)(90 + m) * 4096] = hi;
      }
    }
  }
}

// ---------------------------------------------------------------------------
// WMMA fragment helpers (V_WMMA_F32_16X16X32_F16 layouts, wave32).
// A 16x32: lane&15 = row; lanes 0-15 hold K {0..7,16..23}, lanes 16-31 hold
// K {8..15,24..31}. B 32x16 symmetric: lane&15 = col, same K split.
// All bases/strides are 16B-aligned -> two ds_load_b128 per fragment.
// ---------------------------------------------------------------------------
__device__ inline v16h frag_lds(const _Float16* base, int stride, int row,
                                int k0, int khalf) {
  const v8h* p = (const v8h*)(base + row * stride + k0 + khalf);
  v8h lo = p[0];
  v8h hi = p[2];            // +16 halves
  v16h f;
  #pragma unroll
  for (int i = 0; i < 8; ++i) { f[i] = lo[i]; f[8 + i] = hi[i]; }
  return f;
}

// Unguarded global B fragment (w2: K=128, always in-bounds, 32B-aligned rows).
__device__ inline v16h frag_b_global(const float* __restrict__ wrow,
                                     int k0, int khalf) {
  const v4f* p = (const v4f*)(wrow + k0 + khalf);
  v4f a0 = p[0], a1 = p[1];
  v4f b0 = p[4], b1 = p[5];  // +16 floats
  v16h f;
  #pragma unroll
  for (int i = 0; i < 4; ++i) {
    f[i]      = (_Float16)a0[i];
    f[4 + i]  = (_Float16)a1[i];
    f[8 + i]  = (_Float16)b0[i];
    f[12 + i] = (_Float16)b1[i];
  }
  return f;
}

// ---------------------------------------------------------------------------
// Kernel 2: block = 64 groups. Raw 64*180 f32 (contiguous) -> LDS f16;
// 2x7 conv+ReLU -> A[64][96] f16; FC1 WMMA (K=96, N=128, B=w1 staged in LDS
// pre-padded) +bias+ReLU -> LDS A2[64][128] (aliases dead raw buffer);
// FC2 WMMA (K=128, N=256, B=w2 from global unguarded) +bias -> out.
// 8 waves: wave w = (m-tile w&3, N-half w>>2). Static LDS = 59.9 KB.
// ---------------------------------------------------------------------------
__global__ __launch_bounds__(256) void fused_mlp_kernel(
    const float* __restrict__ cd2,
    const float* __restrict__ conv_w, const float* __restrict__ conv_b,
    const float* __restrict__ w1, const float* __restrict__ b1,
    const float* __restrict__ w2, const float* __restrict__ b2,
    float* __restrict__ out) {
  __shared__ __align__(16) _Float16 s_w1[128 * 96];     // 24576 B, K-padded
  __shared__ __align__(16) _Float16 s_rawa2[64 * 180];  // 23040 B raw / A2 alias
  __shared__ __align__(16) _Float16 s_a[64 * 96];       // 12288 B

  int tid = threadIdx.x;
  int g0  = blockIdx.x * 64;

  // Stage w1 -> LDS f16, zero-pad K 90->96. Branchless: clamp addr + select.
  for (int i = tid; i < 128 * 96; i += 256) {
    int r = i / 96, k = i - r * 96;
    int kc = (k < 90) ? k : 89;
    float v = w1[r * 90 + kc];
    s_w1[i] = (k < 90) ? (_Float16)v : (_Float16)0.f;
  }

  // Stage raw groups -> LDS f16 (64*180 floats are consecutive in cd2).
  const v4f* src4 = (const v4f*)(cd2 + (size_t)g0 * 180);
  for (int i = tid; i < (64 * 180) / 4; i += 256) {
    v4f v = src4[i];
    #pragma unroll
    for (int j = 0; j < 4; ++j) s_rawa2[i * 4 + j] = (_Float16)v[j];
  }
  __syncthreads();

  // 2x7 conv (pad 3 in W) + bias + ReLU -> s_a, zero-padded K to 96.
  // LDS reads are speculatable: clamp index + cndmask, no exec branches.
  float cw[14];
  #pragma unroll
  for (int i = 0; i < 14; ++i) cw[i] = conv_w[i];
  float cb = conv_b[0];
  for (int idx = tid; idx < 64 * 96; idx += 256) {
    int r = idx / 96, k = idx - r * 96;
    float acc = cb;
    #pragma unroll
    for (int j = 0; j < 7; ++j) {
      int wi  = k + j - 3;
      int wic = wi < 0 ? 0 : (wi > 89 ? 89 : wi);
      float v0 = (float)s_rawa2[r * 180 + wic];
      float v1 = (float)s_rawa2[r * 180 + 90 + wic];
      bool ok = (unsigned)wi < 90u;
      acc += ok ? (cw[j] * v0 + cw[7 + j] * v1) : 0.f;
    }
    float res = fmaxf(acc, 0.f);
    s_a[idx] = (k < 90) ? (_Float16)res : (_Float16)0.f;
  }
  __syncthreads();   // s_rawa2 raw contents dead after this point

  int lane  = tid & 31;
  int wv    = tid >> 5;
  int mt    = wv & 3;
  int nh    = wv >> 2;
  int lrow  = lane & 15;
  int khalf = (lane < 16) ? 0 : 8;   // K-offset for A/B, also M-offset for C/D
  int mbase = mt * 16;
  _Float16* s_a2 = s_rawa2;          // reuse as A2[64][128]

  // FC1: ReLU(A @ w1^T + b1) -> s_a2.  K=96 (3 steps), N=128 (2 waves x 4).
  #pragma unroll
  for (int nt = 0; nt < 4; ++nt) {
    int n = (nh * 4 + nt) * 16 + lrow;
    v8f c = {};
    #pragma unroll
    for (int kt = 0; kt < 3; ++kt) {
      v16h a = frag_lds(s_a, 96, mbase + lrow, kt * 32, khalf);
      v16h b = frag_lds(s_w1, 96, n, kt * 32, khalf);
      c = __builtin_amdgcn_wmma_f32_16x16x32_f16(false, a, false, b,
                                                 (short)0, c, false, false);
    }
    float bias = b1[n];
    #pragma unroll
    for (int r = 0; r < 8; ++r)
      s_a2[(mbase + khalf + r) * 128 + n] = (_Float16)fmaxf(c[r] + bias, 0.f);
  }
  __syncthreads();

  // FC2: A2 @ w2^T + b2 -> out.  K=128 (4 steps), N=256 (2 waves x 8).
  #pragma unroll
  for (int nt = 0; nt < 8; ++nt) {
    int n = nh * 128 + nt * 16 + lrow;
    v8f c = {};
    #pragma unroll
    for (int kt = 0; kt < 4; ++kt) {
      v16h a = frag_lds(s_a2, 128, mbase + lrow, kt * 32, khalf);
      v16h b = frag_b_global(w2 + (size_t)n * 128, kt * 32, khalf);
      c = __builtin_amdgcn_wmma_f32_16x16x32_f16(false, a, false, b,
                                                 (short)0, c, false, false);
    }
    float bias = b2[n];
    #pragma unroll
    for (int r = 0; r < 8; ++r)
      out[(size_t)(g0 + mbase + khalf + r) * 256 + n] = c[r] + bias;
  }
}

extern "C" void kernel_launch(void* const* d_in, const int* in_sizes, int n_in,
                              void* d_out, int out_size, void* d_ws, size_t ws_size,
                              hipStream_t stream) {
  const float* x      = (const float*)d_in[0];
  const float* conv_w = (const float*)d_in[1];
  const float* conv_b = (const float*)d_in[2];
  const float* w1     = (const float*)d_in[3];
  const float* b1     = (const float*)d_in[4];
  const float* w2     = (const float*)d_in[5];
  const float* b2     = (const float*)d_in[6];
  float* out = (float*)d_out;
  float* cd2 = (float*)d_ws;   // needs 16*180*64*64*4 = 47,185,920 bytes

  dwt2_kernel<<<256, 256, 0, stream>>>(x, cd2);
  fused_mlp_kernel<<<1024, 256, 0, stream>>>(cd2, conv_w, conv_b, w1, b1, w2, b2, out);
}
